// GCN_flatten_50835232916294
// MI455X (gfx1250) — matrix operand order repaired and checked by reading, not compiled, not run on hardware
//
#include <hip/hip_runtime.h>

// GCN 2-layer forward for MI455X (gfx1250, wave32, WMMA + TDM).
// h1 = (X * normS) @ W1 ; agg1 = scatter_add(h1[src] -> dst) ; x2 = relu(agg1*normD + b1)
// h2 = (x2 * normS) @ W2 ; agg2 = scatter_add(h2[src] -> dst) ; out = agg2*normD + b2

typedef _Float16     v16h  __attribute__((ext_vector_type(16)));
typedef float        v8f   __attribute__((ext_vector_type(8)));
typedef unsigned int u32x4 __attribute__((ext_vector_type(4)));
typedef int          i32x4 __attribute__((ext_vector_type(4)));
typedef int          i32x8 __attribute__((ext_vector_type(8)));

#define IN_F  512
#define HID_F 128
#define OUT_F 40

// ---------------- utility kernels ----------------

__global__ void k_zero4(float4* __restrict__ p, long long n4) {
  long long i = blockIdx.x * (long long)blockDim.x + threadIdx.x;
  if (i < n4) p[i] = make_float4(0.f, 0.f, 0.f, 0.f);
}

__global__ void k_degree(const int* __restrict__ src, const int* __restrict__ dst,
                         float* __restrict__ degOut, float* __restrict__ degIn, int E) {
  int e = blockIdx.x * blockDim.x + threadIdx.x;
  if (e >= E) return;
  atomicAdd(&degOut[src[e]], 1.0f);
  atomicAdd(&degIn[dst[e]], 1.0f);
}

__global__ void k_norm(float* __restrict__ p, int n) {
  int i = blockIdx.x * blockDim.x + threadIdx.x;
  if (i < n) p[i] = rsqrtf(fmaxf(p[i], 1.0f));
}

// Convert W1 (512x128 f32, row-major K x Ncols) into WMMA B-fragment order f16.
// Fragment (kt, ct): 32 lanes x 16 halves, stored contiguously per lane.
// B 32x16 f16 layout: lanes 0-15 -> K=kt*32+0..15, lanes 16-31 -> K=kt*32+16..31,
// halves [2v],[2v+1] = K pair (2v, 2v+1) within the lane group; N = lane&15.
__global__ void k_convW1(const float* __restrict__ W, _Float16* __restrict__ Wf) {
  int t = blockIdx.x * blockDim.x + threadIdx.x;
  if (t >= 16 * 8 * 32) return;
  int lane = t & 31, frag = t >> 5;
  int ct = frag & 7, kt = frag >> 3;
  int n = lane & 15, hsel = lane >> 4;
  _Float16* o = Wf + (size_t)t * 16;
#pragma unroll
  for (int v = 0; v < 8; ++v) {
    int kk = kt * 32 + hsel * 16 + 2 * v;
    o[2 * v]     = (_Float16)W[(size_t)kk * HID_F + ct * 16 + n];
    o[2 * v + 1] = (_Float16)W[(size_t)(kk + 1) * HID_F + ct * 16 + n];
  }
}

// W2 (128x40) -> fragment order, padded to 48 cols (3 col tiles), 4 K tiles.
__global__ void k_convW2(const float* __restrict__ W, _Float16* __restrict__ Wf) {
  int t = blockIdx.x * blockDim.x + threadIdx.x;
  if (t >= 4 * 3 * 32) return;
  int lane = t & 31, frag = t >> 5;
  int ct = frag % 3, kt = frag / 3;
  int n = lane & 15, hsel = lane >> 4;
  int col = ct * 16 + n;
  _Float16* o = Wf + (size_t)t * 16;
#pragma unroll
  for (int v = 0; v < 8; ++v) {
    int kk = kt * 32 + hsel * 16 + 2 * v;
    float a0 = (col < OUT_F) ? W[(size_t)kk * OUT_F + col] : 0.f;
    float a1 = (col < OUT_F) ? W[(size_t)(kk + 1) * OUT_F + col] : 0.f;
    o[2 * v]     = (_Float16)a0;
    o[2 * v + 1] = (_Float16)a1;
  }
}

// ---------------- Layer-1 WMMA GEMM with TDM-staged weights ----------------
// Each block: 8 waves x 16 rows = 128 rows, full 128-col output width.
// W1f (131072 B, fragment order) is DMA'd into LDS once per block by the
// Tensor Data Mover (wave 0 issues; TDM ignores EXEC), then all WMMA B
// operands come from ds_load_b128.
__global__ __launch_bounds__(256) void k_gemm1(
    const float* __restrict__ X, const _Float16* __restrict__ Wf,
    const float* __restrict__ normS, float* __restrict__ H, int nRows) {
  extern __shared__ __align__(32) _Float16 sW[];   // 65536 halves = 131072 B

#if __has_builtin(__builtin_amdgcn_tensor_load_to_lds)
  if (threadIdx.x == 0) {
    unsigned long long ga = (unsigned long long)(uintptr_t)Wf;
    unsigned int ldsOff = (unsigned int)(uintptr_t)sW;  // low 32 bits = LDS byte offset
    // D# group0: count=1 | lds_addr | global_addr[56:0] | type=2 ("image")
    u32x4 g0 = { 1u, ldsOff,
                 (unsigned int)(ga & 0xFFFFFFFFu),
                 (unsigned int)((ga >> 32) & 0x01FFFFFFu) | (2u << 30) };
    // D# group1: data_size=4B; tensor_dim0=tile_dim0=32768 elems (linear 128KB);
    // tensor_dim1=tile_dim1=1; dim0_stride=32768.
    i32x8 g1 = { 0x00020000, (int)0x80000000, 0x00010000, (int)0x80000000,
                 1, 32768, 0, 0 };
    i32x4 gz = { 0, 0, 0, 0 };
#if defined(__clang_major__) && (__clang_major__ >= 23)
    i32x8 gz8 = { 0, 0, 0, 0, 0, 0, 0, 0 };
    __builtin_amdgcn_tensor_load_to_lds(g0, g1, gz, gz, gz8, 0);
#else
    __builtin_amdgcn_tensor_load_to_lds(g0, g1, gz, gz, 0);
#endif
    __builtin_amdgcn_s_wait_tensorcnt(0);
  }
#else
  // Fallback: cooperative copy (16B per thread per step).
  {
    const uint4* gsrc = (const uint4*)Wf;
    uint4* ldst = (uint4*)sW;
    for (int i = threadIdx.x; i < 131072 / 16; i += 256) ldst[i] = gsrc[i];
  }
#endif
  __syncthreads();

  int lane = threadIdx.x & 31;
  int wave = threadIdx.x >> 5;
  int rowBase = blockIdx.x * 128 + wave * 16;
  int m = lane & 15;
  int hsel = lane >> 4;
  int row = rowBase + m;
  int rowc = row < nRows ? row : nRows - 1;  // clamp loads; stores masked (EXEC all-1 for WMMA)
  float ns = normS[rowc];
  const float* xrow = X + (size_t)rowc * IN_F;

  v8f acc[8];
#pragma unroll
  for (int ct = 0; ct < 8; ++ct) {
    v8f z = {0.f, 0.f, 0.f, 0.f, 0.f, 0.f, 0.f, 0.f};
    acc[ct] = z;
  }

  for (int kt = 0; kt < 16; ++kt) {
    // A fragment 16x32 f16: lane m holds row m; K pairs per VGPR,
    // lanes 0-15 K=0..15, lanes 16-31 K=16..31. Fused normS scaling + f32->f16 convert.
    v16h a;
#pragma unroll
    for (int v = 0; v < 8; ++v) {
      int kk = kt * 32 + hsel * 16 + 2 * v;
      float2 xv = *(const float2*)(xrow + kk);
      a[2 * v]     = (_Float16)(xv.x * ns);
      a[2 * v + 1] = (_Float16)(xv.y * ns);
    }
#pragma unroll
    for (int ct = 0; ct < 8; ++ct) {
      v16h b = *(const v16h*)(sW + ((size_t)(kt * 8 + ct) * 32 + lane) * 16);
      acc[ct] = __builtin_amdgcn_wmma_f32_16x16x32_f16(
          false, a, false, b, (short)0, acc[ct], false, false);
    }
  }

  // C/D layout: VGPR j -> M=j (lanes 0-15) / M=j+8 (lanes 16-31); N = lane&15.
#pragma unroll
  for (int j = 0; j < 8; ++j) {
    int r = rowBase + j + hsel * 8;
    if (r < nRows) {
      float* orow = H + (size_t)r * HID_F;
#pragma unroll
      for (int ct = 0; ct < 8; ++ct) orow[ct * 16 + m] = acc[ct][j];
    }
  }
}

// ---------------- Generic WMMA GEMM (layer 2; B from global/L2) ----------------
template <int KT, int NCT>
__global__ __launch_bounds__(256) void k_gemm(
    const float* __restrict__ X, const _Float16* __restrict__ Wf,
    const float* __restrict__ normS, float* __restrict__ H,
    int nRows, int outStride, int outCols) {
  const int K = KT * 32;
  int lane = threadIdx.x & 31;
  int wave = threadIdx.x >> 5;
  int rowBase = blockIdx.x * 128 + wave * 16;
  int m = lane & 15;
  int hsel = lane >> 4;
  int row = rowBase + m;
  int rowc = row < nRows ? row : nRows - 1;
  float ns = normS[rowc];
  const float* xrow = X + (size_t)rowc * K;

  v8f acc[NCT];
#pragma unroll
  for (int ct = 0; ct < NCT; ++ct) {
    v8f z = {0.f, 0.f, 0.f, 0.f, 0.f, 0.f, 0.f, 0.f};
    acc[ct] = z;
  }

  for (int kt = 0; kt < KT; ++kt) {
    v16h a;
#pragma unroll
    for (int v = 0; v < 8; ++v) {
      int kk = kt * 32 + hsel * 16 + 2 * v;
      float2 xv = *(const float2*)(xrow + kk);
      a[2 * v]     = (_Float16)(xv.x * ns);
      a[2 * v + 1] = (_Float16)(xv.y * ns);
    }
#pragma unroll
    for (int ct = 0; ct < NCT; ++ct) {
      v16h b = *(const v16h*)(Wf + ((size_t)(kt * NCT + ct) * 32 + lane) * 16);
      acc[ct] = __builtin_amdgcn_wmma_f32_16x16x32_f16(
          false, a, false, b, (short)0, acc[ct], false, false);
    }
  }

#pragma unroll
  for (int j = 0; j < 8; ++j) {
    int r = rowBase + j + hsel * 8;
    if (r < nRows) {
      float* orow = H + (size_t)r * outStride;
#pragma unroll
      for (int ct = 0; ct < NCT; ++ct) {
        int col = ct * 16 + m;
        if (col < outCols) orow[col] = acc[ct][j];
      }
    }
  }
}

// ---------------- SpMM scatter-add (edge-parallel float atomics, L2-resident) ----------------

__global__ void k_agg128(const float* __restrict__ H, const int* __restrict__ src,
                         const int* __restrict__ dst, float* __restrict__ agg, int E) {
  long long idx = blockIdx.x * (long long)blockDim.x + threadIdx.x;
  int e = (int)(idx >> 5);
  if (e >= E) return;
  int c0 = ((int)idx & 31) * 4;
  int s = src[e], d = dst[e];
  const float4 v = *(const float4*)(H + (size_t)s * HID_F + c0);
  float* o = agg + (size_t)d * HID_F + c0;
  atomicAdd(o + 0, v.x);
  atomicAdd(o + 1, v.y);
  atomicAdd(o + 2, v.z);
  atomicAdd(o + 3, v.w);
}

__global__ void k_agg40(const float* __restrict__ H, const int* __restrict__ src,
                        const int* __restrict__ dst, float* __restrict__ agg, int E) {
  long long idx = blockIdx.x * (long long)blockDim.x + threadIdx.x;
  int e = (int)(idx >> 3);
  if (e >= E) return;
  int c0 = ((int)idx & 7) * 5;
  int s = src[e], d = dst[e];
  const float* hrow = H + (size_t)s * OUT_F + c0;
  float* o = agg + (size_t)d * OUT_F + c0;
#pragma unroll
  for (int i = 0; i < 5; ++i) atomicAdd(o + i, hrow[i]);
}

// ---------------- epilogues ----------------

__global__ void k_post1(const float* __restrict__ agg, const float* __restrict__ normD,
                        const float* __restrict__ bias, float* __restrict__ X2, long long total) {
  long long idx = blockIdx.x * (long long)blockDim.x + threadIdx.x;
  if (idx >= total) return;
  int i = (int)(idx >> 7);
  int c = (int)idx & 127;
  float v = agg[idx] * normD[i] + bias[c];
  X2[idx] = v > 0.f ? v : 0.f;
}

__global__ void k_post2(const float* __restrict__ agg, const float* __restrict__ normD,
                        const float* __restrict__ bias, float* __restrict__ out, long long total) {
  long long idx = blockIdx.x * (long long)blockDim.x + threadIdx.x;
  if (idx >= total) return;
  int i = (int)(idx / OUT_F);
  int c = (int)(idx - (long long)i * OUT_F);
  out[idx] = agg[idx] * normD[i] + bias[c];
}

// ---------------- driver ----------------

extern "C" void kernel_launch(void* const* d_in, const int* in_sizes, int n_in,
                              void* d_out, int out_size, void* d_ws, size_t ws_size,
                              hipStream_t stream) {
  (void)n_in; (void)out_size; (void)ws_size;
  const float* features = (const float*)d_in[0];
  const int*   src      = (const int*)d_in[1];
  const int*   dst      = (const int*)d_in[2];
  const float* W1       = (const float*)d_in[3];
  const float* b1       = (const float*)d_in[4];
  const float* W2       = (const float*)d_in[5];
  const float* b2       = (const float*)d_in[6];
  float* out = (float*)d_out;

  const int N = in_sizes[0] / IN_F;   // 100000
  const int E = in_sizes[1];          // 1600000

  // Workspace (floats): [normS(N) normD(N) agg1(128N) agg2(40N) h1(128N) h2(40N)] [W1f f16][W2f f16]
  float* wsf   = (float*)d_ws;
  float* normS = wsf;
  float* normD = wsf + N;
  float* agg1  = wsf + 2ll * N;
  float* agg2  = agg1 + (size_t)HID_F * N;
  float* h1    = agg2 + (size_t)OUT_F * N;      // reused as x2 after post1
  float* h2    = h1 + (size_t)HID_F * N;
  _Float16* W1f = (_Float16*)(h2 + (size_t)OUT_F * N);
  _Float16* W2f = W1f + (size_t)IN_F * HID_F;

  const int B = 256;

  long long zf = (2ll + HID_F + OUT_F) * N;
  long long n4 = zf / 4;
  k_zero4<<<(unsigned)((n4 + B - 1) / B), B, 0, stream>>>((float4*)wsf, n4);

  k_degree<<<(E + B - 1) / B, B, 0, stream>>>(src, dst, normS, normD, E);
  k_norm<<<(2 * N + B - 1) / B, B, 0, stream>>>(normS, 2 * N);

  k_convW1<<<(16 * 8 * 32 + B - 1) / B, B, 0, stream>>>(W1, W1f);
  k_convW2<<<(4 * 3 * 32 + B - 1) / B, B, 0, stream>>>(W2, W2f);

  int gBlocks = (N + 127) / 128;

  // Layer 1: TDM-staged WMMA GEMM -> scatter-add -> norm+bias+relu.
  k_gemm1<<<gBlocks, B, 131072, stream>>>(features, W1f, normS, h1, N);
  long long t1 = (long long)E * 32;
  k_agg128<<<(unsigned)((t1 + B - 1) / B), B, 0, stream>>>(h1, src, dst, agg1, E);
  long long t2 = (long long)N * HID_F;
  k_post1<<<(unsigned)((t2 + B - 1) / B), B, 0, stream>>>(agg1, normD, b1, h1, t2);

  // Layer 2: WMMA GEMM (48 padded cols, masked store to 40) -> scatter-add -> norm+bias.
  k_gemm<4, 3><<<gBlocks, B, 0, stream>>>(h1, W2f, normS, h2, N, OUT_F, OUT_F);
  long long t3 = (long long)E * 8;
  k_agg40<<<(unsigned)((t3 + B - 1) / B), B, 0, stream>>>(h2, src, dst, agg2, E);
  long long t4 = (long long)N * OUT_F;
  k_post2<<<(unsigned)((t4 + B - 1) / B), B, 0, stream>>>(agg2, normD, b2, out, t4);
}
